// ASAP_Pooling_38414187496034
// MI455X (gfx1250) — matrix operand order repaired and checked by reading, not compiled, not run on hardware
//
#include <hip/hip_runtime.h>

#define NN 4096          // nodes
#define CC 256           // channels
#define EE0 65536        // edges before self loops
#define EE (EE0 + NN)    // edges incl self loops
#define KK 2048          // pooled nodes
#define NEG_SLOPE 0.2f

typedef __attribute__((ext_vector_type(16))) __bf16 v16bf;
typedef __attribute__((ext_vector_type(8)))  float  v8f;

__device__ __forceinline__ unsigned short f2bf(float f) {
  union { float f; unsigned u; } x; x.f = f;
  unsigned u = x.u;
  unsigned r = u + 0x7FFFu + ((u >> 16) & 1u);   // round-to-nearest-even
  return (unsigned short)(r >> 16);
}

// ---------------- CSR construction ----------------
__global__ void k_deg(const int* ei, int* deg) {
  int e = blockIdx.x * 256 + threadIdx.x;
  if (e >= EE) return;
  int i = (e < EE0) ? ei[e] : (e - EE0);
  atomicAdd(&deg[i], 1);
}

__global__ void __launch_bounds__(1024) k_scan(const int* deg, int* rowstart) {
  __shared__ int s[1024];
  int t = threadIdx.x;
  int base = t * 4;
  int v0 = deg[base], v1 = deg[base + 1], v2 = deg[base + 2], v3 = deg[base + 3];
  int sum = v0 + v1 + v2 + v3;
  s[t] = sum; __syncthreads();
  for (int off = 1; off < 1024; off <<= 1) {
    int v = (t >= off) ? s[t - off] : 0;
    __syncthreads();
    s[t] += v;
    __syncthreads();
  }
  int excl = s[t] - sum;
  rowstart[base]     = excl;
  rowstart[base + 1] = excl + v0;
  rowstart[base + 2] = excl + v0 + v1;
  rowstart[base + 3] = excl + v0 + v1 + v2;
  if (t == 1023) rowstart[NN] = s[t];
}

__global__ void k_fill(const int* ei, const int* rowstart, int* cursor,
                       int* col, int* eid) {
  int e = blockIdx.x * 256 + threadIdx.x;
  if (e >= EE) return;
  int i, j;
  if (e < EE0) { i = ei[e]; j = ei[EE0 + e]; } else { i = e - EE0; j = i; }
  int p = rowstart[i] + atomicAdd(&cursor[i], 1);
  col[p] = j;
  eid[p] = e;
}

// make CSR canonical (sorted by original edge id) -> deterministic fp order
__global__ void k_sortrow(const int* rowstart, int* col, int* eid) {
  int n = blockIdx.x * 256 + threadIdx.x;
  if (n >= NN) return;
  int rs = rowstart[n], re = rowstart[n + 1];
  for (int p = rs + 1; p < re; ++p) {
    int ke = eid[p], kc = col[p];
    int q = p - 1;
    while (q >= rs && eid[q] > ke) { eid[q + 1] = eid[q]; col[q + 1] = col[q]; --q; }
    eid[q + 1] = ke; col[q + 1] = kc;
  }
}

// ---------------- attention ----------------
__global__ void k_segmax(const float* x, const int* rowstart, const int* col,
                         float* Xq) {
  int n = blockIdx.x, c = threadIdx.x;
  int rs = rowstart[n], re = rowstart[n + 1];
  float m = -INFINITY;
  for (int p = rs; p < re; ++p)
    m = fmaxf(m, x[(size_t)col[p] * CC + c]);
  Xq[(size_t)n * CC + c] = m;
}

// fold Mq@w1 into a matvec: wq_eff = Wq @ gat_w[:C], qconst = bq.w1 + gat_b
__global__ void k_prep(const float* Wq, const float* bq, const float* gw,
                       const float* gb, float* wq_eff, float* qc) {
  int k = threadIdx.x;
  float acc = 0.f;
  for (int c = 0; c < CC; ++c) acc += Wq[k * CC + c] * gw[c];
  wq_eff[k] = acc;
  if (k == 0) {
    float s = gb[0];
    for (int c = 0; c < CC; ++c) s += bq[c] * gw[c];
    qc[0] = s;
  }
}

__global__ void k_qk(const float* Xq, const float* x, const float* wq_eff,
                     const float* qc, const float* gw, float* qv, float* kv) {
  int wave = threadIdx.x >> 5, lane = threadIdx.x & 31;
  int n = blockIdx.x * 8 + wave;
  const float* w2 = gw + CC;
  float aq = 0.f, ak = 0.f;
  for (int c = lane; c < CC; c += 32) {
    aq += Xq[(size_t)n * CC + c] * wq_eff[c];
    ak += x[(size_t)n * CC + c] * w2[c];
  }
  for (int off = 16; off; off >>= 1) {
    aq += __shfl_down(aq, off, 32);
    ak += __shfl_down(ak, off, 32);
  }
  if (lane == 0) { qv[n] = aq + qc[0]; kv[n] = ak; }
}

// fused per-row: leakyrelu -> segment softmax -> out[n] = sum score*x[j]
__global__ void k_softrow(const float* x, const int* rowstart, const int* col,
                          const float* qv, const float* kv, float* score,
                          float* out) {
  __shared__ float red[256];
  __shared__ float s_sc[256];
  __shared__ int   s_col[256];
  int n = blockIdx.x, t = threadIdx.x;
  int rs = rowstart[n], d = rowstart[n + 1] - rs;
  float qn = qv[n];

  float lm = -INFINITY;
  for (int e = t; e < d; e += 256) {
    float sc = qn + kv[col[rs + e]];
    sc = sc > 0.f ? sc : NEG_SLOPE * sc;
    lm = fmaxf(lm, sc);
  }
  red[t] = lm; __syncthreads();
  for (int off = 128; off; off >>= 1) {
    if (t < off) red[t] = fmaxf(red[t], red[t + off]);
    __syncthreads();
  }
  float m = red[0]; __syncthreads();

  float lz = 0.f;
  for (int e = t; e < d; e += 256) {
    float sc = qn + kv[col[rs + e]];
    sc = sc > 0.f ? sc : NEG_SLOPE * sc;
    lz += __expf(sc - m);
  }
  red[t] = lz; __syncthreads();
  for (int off = 128; off; off >>= 1) {
    if (t < off) red[t] += red[t + off];
    __syncthreads();
  }
  float inv = 1.f / red[0]; __syncthreads();

  float acc = 0.f;                       // thread t owns channel t
  for (int base = 0; base < d; base += 256) {
    int e = base + t;
    if (e < d) {
      int cj = col[rs + e];
      float sc = qn + kv[cj];
      sc = sc > 0.f ? sc : NEG_SLOPE * sc;
      float s = __expf(sc - m) * inv;
      score[rs + e] = s;
      s_sc[t] = s; s_col[t] = cj;
    }
    __syncthreads();
    int lim = min(256, d - base);
    for (int u = 0; u < lim; ++u)
      acc += s_sc[u] * x[(size_t)s_col[u] * CC + t];
    __syncthreads();
  }
  out[(size_t)n * CC + t] = acc;
}

// ---------------- LEConv fitness ----------------
__global__ void k_ledots(const float* out, const float* le1_w, const float* le1_b,
                         const float* le2_w, const float* le3_w,
                         float* av, float* bv, float* l3v) {
  int wave = threadIdx.x >> 5, lane = threadIdx.x & 31;
  int n = blockIdx.x * 8 + wave;
  float a = 0.f, b = 0.f, c3 = 0.f;
  for (int c = lane; c < CC; c += 32) {
    float o = out[(size_t)n * CC + c];
    a += o * le1_w[c]; b += o * le2_w[c]; c3 += o * le3_w[c];
  }
  for (int off = 16; off; off >>= 1) {
    a += __shfl_down(a, off, 32);
    b += __shfl_down(b, off, 32);
    c3 += __shfl_down(c3, off, 32);
  }
  if (lane == 0) { av[n] = a + le1_b[0]; bv[n] = b; l3v[n] = c3; }
}

__global__ void k_fit(const int* rowstart, const int* col, const float* av,
                      const float* bv, const float* l3v, const float* le3_b,
                      float* fit) {
  int wave = threadIdx.x >> 5, lane = threadIdx.x & 31;
  int n = blockIdx.x * 8 + wave;
  int rs = rowstart[n], d = rowstart[n + 1] - rs;
  float s = 0.f;
  for (int e = lane; e < d; e += 32) s += av[col[rs + e]];
  for (int off = 16; off; off >>= 1) s += __shfl_down(s, off, 32);
  if (lane == 0) {
    float z = s - (float)d * bv[n] + l3v[n] + le3_b[0];
    fit[n] = 1.f / (1.f + __expf(-z));
  }
}

// ---------------- top-K by exact rank (stable, matches lax.top_k) ----------
__global__ void k_rank(const float* fit, int* perm, int* nidx, int* kept) {
  int n = blockIdx.x * 256 + threadIdx.x;
  float f = fit[n];
  int r = 0;
  for (int m = 0; m < NN; ++m) {
    float g = fit[m];
    r += (g > f) || (g == f && m < n);
  }
  if (r < KK) { perm[r] = n; nidx[n] = r; kept[n] = 1; }
  else kept[n] = 0;
}

__global__ void k_xout(const int* perm, const float* fit, const float* out,
                       float* dout) {
  int r = blockIdx.x, c = threadIdx.x;
  int p = perm[r];
  float f = fit[p];
  dout[(size_t)r * CC + c] = out[(size_t)p * CC + c] * f;
  if (c == 0)  // perm tail (as float, concatenated output convention)
    dout[(size_t)KK * CC + (size_t)KK * KK + r] = (float)p;
}

// ---------------- connectivity: Eadj = S^T (A S) ----------------
// S scatter: one thread per row -> each (j,cluster) address written by one
// thread in a fixed order => deterministic.
__global__ void k_scatS(const int* rowstart, const int* col, const float* score,
                        const int* kept, const int* nidx, float* S) {
  int n = blockIdx.x * 256 + threadIdx.x;
  if (n >= NN || !kept[n]) return;
  int rs = rowstart[n], re = rowstart[n + 1];
  int cidx = nidx[n];
  for (int p = rs; p < re; ++p)
    atomicAdd(&S[(size_t)col[p] * KK + cidx], score[p]);
}

// T = A @ S row-wise (no atomics), written as bf16 in k-pair-interleaved
// layout: element (k, n) lives at ushort index ((k>>1)*KK + n)*2 + (k&1),
// so the WMMA B fragment's (k, k+1) pair is one aligned 32-bit load.
__global__ void k_T(const int* rowstart, const int* col, const float* S,
                    unsigned short* Tb) {
  int n = blockIdx.x, t = threadIdx.x;
  int rs = rowstart[n], re = rowstart[n + 1];
  size_t base = (size_t)(n >> 1) * (2 * KK) + (n & 1);
  for (int cb = 0; cb < KK; cb += 256) {
    float acc = 0.f;
    for (int p = rs; p < re; ++p)
      acc += S[(size_t)col[p] * KK + cb + t];
    Tb[base + (size_t)(cb + t) * 2] = f2bf(acc);
  }
}

// St = S^T as bf16 [KK x NN] via LDS tile transpose
__global__ void k_transS(const float* S, unsigned short* St) {
  __shared__ float tile[32][33];
  int rb = blockIdx.x * 32;  // pooled-cluster dim (KK)
  int jb = blockIdx.y * 32;  // node dim (NN)
  for (int rr = threadIdx.y; rr < 32; rr += 8)
    tile[rr][threadIdx.x] = S[(size_t)(jb + rr) * KK + rb + threadIdx.x];
  __syncthreads();
  for (int rr = threadIdx.y; rr < 32; rr += 8)
    St[(size_t)(rb + rr) * NN + jb + threadIdx.x] =
        f2bf(tile[threadIdx.x][rr]);
}

// Eadj[KKxKK] = St[KKxNN] @ T[NNxKK], bf16 WMMA, f32 accum.
// Each wave computes a 16x64 output strip: one A fragment feeds 4 WMMAs.
// A: two contiguous 16B runs per lane -> 2x global_load_b128.
// B: pair-interleaved layout -> 8x coalesced global_load_b32 per fragment.
__global__ void __launch_bounds__(256) k_gemm(const unsigned short* St,
                                              const unsigned short* Tb,
                                              float* Eadj) {
  int wave = threadIdx.x >> 5, lane = threadIdx.x & 31;
  int tile = blockIdx.x * 8 + wave;
  int tr  = tile >> 5;          // 128 row tiles of 16
  int tcs = tile & 31;          // 32 col strips of 64
  int row0 = tr * 16, col0 = tcs * 64;
  int m = lane & 15, half = lane >> 4;

  v8f c[4] = {{}, {}, {}, {}};
  const unsigned short* arow = St + (size_t)(row0 + m) * NN;

  for (int k0 = 0; k0 < NN; k0 += 32) {
    union { v16bf v; unsigned u[8]; } A;
#pragma unroll
    for (int vr = 0; vr < 8; ++vr) {
      // ISA 7.12.2: 16-bit A 16x32 — VGPR0-3: K=2v+{0,1}(+8*half), VGPR4-7: +16
      int kp = (vr < 4) ? (2 * vr + 8 * half) : (16 + 2 * (vr - 4) + 8 * half);
      A.u[vr] = *(const unsigned*)(arow + k0 + kp);
    }
#pragma unroll
    for (int s = 0; s < 4; ++s) {
      union { v16bf v; unsigned u[8]; } B;
#pragma unroll
      for (int vr = 0; vr < 8; ++vr) {
        // B VGPR v: K = 2v+{0,1}(+8*half), v>=4: +8; k-pair index = k0/2 + 4*half + v (+4 if v>=4)
        size_t pair = (size_t)(k0 >> 1) + 4 * half + vr + (vr >= 4 ? 4 : 0);
        B.u[vr] = *(const unsigned*)(Tb + (pair * KK + col0 + s * 16 + m) * 2);
      }
      c[s] = __builtin_amdgcn_wmma_f32_16x16x32_bf16(false, A.v, false, B.v,
                                                     (short)0, c[s], false, false);
    }
  }
#pragma unroll
  for (int s = 0; s < 4; ++s)
#pragma unroll
    for (int r = 0; r < 8; ++r) {
      int mo = r + 8 * half;   // C/D layout: VGPR r -> M = r (+8 for hi half)
      Eadj[(size_t)(row0 + mo) * KK + col0 + s * 16 + m] = c[s][r];
    }
}

// diag(Eadj) := 1  (remove_self_loops + add_remaining_self_loops(fill=1))
__global__ void k_diag(float* Eadj) {
  int r = blockIdx.x * 256 + threadIdx.x;
  Eadj[(size_t)r * KK + r] = 1.f;
}

// ---------------- launcher ----------------
static inline size_t al256(size_t x) { return (x + 255) & ~(size_t)255; }

extern "C" void kernel_launch(void* const* d_in, const int* in_sizes, int n_in,
                              void* d_out, int out_size, void* d_ws,
                              size_t ws_size, hipStream_t stream) {
  (void)in_sizes; (void)n_in; (void)out_size; (void)ws_size;
  const float* x     = (const float*)d_in[0];
  const int*   ei    = (const int*)d_in[1];
  const float* Wq    = (const float*)d_in[2];
  const float* bq    = (const float*)d_in[3];
  const float* gatw  = (const float*)d_in[4];
  const float* gatb  = (const float*)d_in[5];
  const float* le1w  = (const float*)d_in[6];
  const float* le1b  = (const float*)d_in[7];
  const float* le2w  = (const float*)d_in[8];
  const float* le3w  = (const float*)d_in[9];
  const float* le3b  = (const float*)d_in[10];
  float* dout = (float*)d_out;
  float* Eadj = dout + (size_t)KK * CC;

  char* w = (char*)d_ws;
  size_t off = 0;
  auto take = [&](size_t bytes) { void* p = w + off; off = al256(off + bytes); return p; };
  int*   deg      = (int*)take(NN * 4);
  int*   rowstart = (int*)take((NN + 1) * 4);
  int*   cursor   = (int*)take(NN * 4);
  int*   csr_col  = (int*)take((size_t)EE * 4);
  int*   csr_eid  = (int*)take((size_t)EE * 4);
  float* Xq       = (float*)take((size_t)NN * CC * 4);
  float* wq_eff   = (float*)take(CC * 4);
  float* qc       = (float*)take(4);
  float* qv       = (float*)take(NN * 4);
  float* kv       = (float*)take(NN * 4);
  float* score    = (float*)take((size_t)EE * 4);
  float* outb     = (float*)take((size_t)NN * CC * 4);
  float* av       = (float*)take(NN * 4);
  float* bv       = (float*)take(NN * 4);
  float* l3v      = (float*)take(NN * 4);
  float* fit      = (float*)take(NN * 4);
  int*   perm     = (int*)take(KK * 4);
  int*   nidx     = (int*)take(NN * 4);
  int*   kept     = (int*)take(NN * 4);
  float* S        = (float*)take((size_t)NN * KK * 4);             // 32 MB
  unsigned short* St = (unsigned short*)take((size_t)KK * NN * 2); // 16 MB
  unsigned short* Tb = (unsigned short*)take((size_t)NN * KK * 2); // 16 MB

  hipMemsetAsync(deg, 0, NN * 4, stream);
  hipMemsetAsync(cursor, 0, NN * 4, stream);
  hipMemsetAsync(S, 0, (size_t)NN * KK * 4, stream);

  k_deg<<<(EE + 255) / 256, 256, 0, stream>>>(ei, deg);
  k_scan<<<1, 1024, 0, stream>>>(deg, rowstart);
  k_fill<<<(EE + 255) / 256, 256, 0, stream>>>(ei, rowstart, cursor, csr_col, csr_eid);
  k_sortrow<<<NN / 256, 256, 0, stream>>>(rowstart, csr_col, csr_eid);
  k_segmax<<<NN, CC, 0, stream>>>(x, rowstart, csr_col, Xq);
  k_prep<<<1, CC, 0, stream>>>(Wq, bq, gatw, gatb, wq_eff, qc);
  k_qk<<<NN / 8, 256, 0, stream>>>(Xq, x, wq_eff, qc, gatw, qv, kv);
  k_softrow<<<NN, CC, 0, stream>>>(x, rowstart, csr_col, qv, kv, score, outb);
  k_ledots<<<NN / 8, 256, 0, stream>>>(outb, le1w, le1b, le2w, le3w, av, bv, l3v);
  k_fit<<<NN / 8, 256, 0, stream>>>(rowstart, csr_col, av, bv, l3v, le3b, fit);
  k_rank<<<NN / 256, 256, 0, stream>>>(fit, perm, nidx, kept);
  k_xout<<<KK, CC, 0, stream>>>(perm, fit, outb, dout);
  k_scatS<<<NN / 256, 256, 0, stream>>>(rowstart, csr_col, score, kept, nidx, S);
  k_T<<<NN, 256, 0, stream>>>(rowstart, csr_col, S, Tb);
  {
    dim3 grid(KK / 32, NN / 32), blk(32, 8);
    k_transS<<<grid, blk, 0, stream>>>(S, St);
  }
  k_gemm<<<(KK / 16) * (KK / 64) / 8, 256, 0, stream>>>(St, Tb, Eadj);
  k_diag<<<KK / 256, 256, 0, stream>>>(Eadj);
}
